// CRF_BiLSTM_87926570483972
// MI455X (gfx1250) — compile-verified, matrix operand order checked by reading, not run on hardware
//
#include <hip/hip_runtime.h>
#include <hip/hip_bf16.h>

// ---------------------------------------------------------------------------
// BiLSTM + CRF (Viterbi) for MI455X / gfx1250, wave32 + WMMA f16->f32.
//
// Shapes: VOCAB=50000, EMB=256, HID=256, TAGS=10, B=32, S=256.
// Strategy:
//   - Hoist the input projection xt@Wih.T for every timestep into one big
//     parallel GEMM [8192,256]@[256,1024] per direction (WMMA f16, f32 acc).
//   - Sequential part: only h@Whh.T (32x1024x256 per step). One persistent
//     workgroup of 16 waves PER DIRECTION, both directions concurrent
//     (2-block launch -> 2 WGPs), h kept in LDS (f16), c kept in WMMA
//     accumulator registers across all 256 steps.
//   - Output projection (N=10) and Viterbi are tiny; scalar VALU.
// ---------------------------------------------------------------------------

typedef __attribute__((ext_vector_type(16))) _Float16 v16h;
typedef __attribute__((ext_vector_type(8)))  _Float16 v8h;
typedef __attribute__((ext_vector_type(8)))  float    v8f;

#define NEG_INF (-10000.0f)
#define SOS_IDX 1
#define EOS_IDX 2

// ---- WMMA helpers ---------------------------------------------------------

__device__ __forceinline__ v8f wmma32(v16h a, v16h b, v8f c) {
  // D = A(16x32 f16) * B(32x16 f16) + C(16x16 f32)
  return __builtin_amdgcn_wmma_f32_16x16x32_f16(
      /*neg_a=*/false, a, /*neg_b=*/false, b,
      /*c_mod=*/(short)0, c, /*reuse_a=*/false, /*reuse_b=*/false);
}

// Load a 16x32 f16 fragment (A layout; also used for B since B[k][n]=W[n][k]
// makes the per-lane addresses identical: lane&15 selects the tile row/col,
// lane>>4 selects which 8-wide K chunk). Per the ISA layout, each lane needs
// two contiguous 8xf16 (16B) chunks: K = half*8..+7 and K = 16+half*8..+7.
__device__ __forceinline__ v16h load_frag(const _Float16* base, int tile_row,
                                          int stride, int k0, int lane) {
  int nn = lane & 15;
  int half = lane >> 4;
  const _Float16* p = base + (size_t)(tile_row + nn) * stride + k0 + half * 8;
  v8h lo = *(const v8h*)(p);
  v8h hi = *(const v8h*)(p + 16);
  return __builtin_shufflevector(lo, hi, 0, 1, 2, 3, 4, 5, 6, 7,
                                 8, 9, 10, 11, 12, 13, 14, 15);
}

__device__ __forceinline__ float sigm(float x) {
  return 1.0f / (1.0f + __expf(-x));
}

// ---- Kernel 1: embedding gather + cast to f16, layout [S][B][E] -----------

__global__ void k_embed_cast(const int* __restrict__ x,     // [B=32][S=256]
                             const float* __restrict__ emb, // [VOCAB][256]
                             _Float16* __restrict__ e16)    // [S*B][256]
{
  size_t i = (size_t)blockIdx.x * 256 + threadIdx.x;  // 8192*256 total
  int col = (int)(i & 255);
  int row = (int)(i >> 8);       // row = s*32 + b
  int s = row >> 5;
  int b = row & 31;
  int tok = x[b * 256 + s];
  e16[i] = (_Float16)emb[(size_t)tok * 256 + col];
}

// ---- Kernel 2: cast 4 weight matrices to f16, fuse biases -----------------

__global__ void k_cast_weights(const float* __restrict__ Wih_f,
                               const float* __restrict__ Whh_f,
                               const float* __restrict__ bih_f,
                               const float* __restrict__ bhh_f,
                               const float* __restrict__ Wih_b,
                               const float* __restrict__ Whh_b,
                               const float* __restrict__ bih_b,
                               const float* __restrict__ bhh_b,
                               _Float16* __restrict__ w16,   // 4 x [1024][256]
                               float* __restrict__ bias)     // 2 x [1024]
{
  int i = blockIdx.x * 256 + threadIdx.x;   // 4*262144 total
  int mi = i >> 18;
  int off = i & 262143;
  const float* src = (mi == 0) ? Wih_f : (mi == 1) ? Whh_f
                   : (mi == 2) ? Wih_b : Whh_b;
  w16[i] = (_Float16)src[off];
  if (i < 2048) {
    bias[i] = (i < 1024) ? (bih_f[i] + bhh_f[i])
                         : (bih_b[i - 1024] + bhh_b[i - 1024]);
  }
}

// ---- Kernel 3: input GEMM  G[dir] = e @ Wih[dir].T + bias -----------------
// M = S*B = 8192, N = 4H = 1024, K = 256. One 16x16 tile per wave,
// 8 x wmma_f32_16x16x32_f16 per tile. grid.x = 8192 (4096 blocks per dir).

__global__ __launch_bounds__(256) void k_input_gemm(
    const _Float16* __restrict__ e16,   // [8192][256]
    const _Float16* __restrict__ w16,   // Wih_f @0, Wih_b @ 2*262144
    const float* __restrict__ bias,     // [2][1024]
    float* __restrict__ G)              // [2][8192][1024]
{
  int w = threadIdx.x >> 5;
  int lane = threadIdx.x & 31;
  int blk = blockIdx.x;
  int dir = blk >> 12;
  blk &= 4095;
  int tile = blk * 8 + w;          // 0..32767
  int mtile = tile >> 6;           // 0..511
  int ntile = tile & 63;           // 0..63

  const _Float16* Wih = w16 + (size_t)dir * 2 * 262144;
  const float* bp = bias + dir * 1024;
  float* Gp = G + (size_t)dir * 8192 * 1024;

  v8f acc = {};
#pragma unroll
  for (int k0 = 0; k0 < 256; k0 += 32) {
    v16h a = load_frag(e16, mtile * 16, 256, k0, lane);
    v16h b = load_frag(Wih, ntile * 16, 256, k0, lane);
    if (k0 + 32 < 256) {  // prefetch next K-chunk of the weight tile
      __builtin_prefetch(Wih + (size_t)(ntile * 16 + (lane & 15)) * 256 +
                             k0 + 32, 0, 1);
    }
    acc = wmma32(a, b, acc);
  }

  int nn = lane & 15;
  int half = lane >> 4;
  float bb = bp[ntile * 16 + nn];
#pragma unroll
  for (int r = 0; r < 8; ++r) {
    int m = mtile * 16 + half * 8 + r;
    Gp[(size_t)m * 1024 + ntile * 16 + nn] = acc[r] + bb;
  }
}

// ---- Kernel 4: persistent recurrent LSTM, both directions concurrently ----
// grid.x = 2 (block 0 = forward, block 1 = backward), 512 threads = 16 waves.
// Wave w: m-tile (w>>3), hidden tiles {w&7, (w&7)+8}. Per step:
// R = h @ Whh.T via WMMA; gates = G_in + R; c kept in accumulator registers,
// h round-trips through LDS as f16.

__device__ __forceinline__ void gate_mm(const float* __restrict__ Grow,
                                        const _Float16* __restrict__ Whh,
                                        const _Float16* __restrict__ hsh,
                                        int mt, int ht, int lane,
                                        v8f& ai, v8f& af, v8f& ag, v8f& ao) {
  int nn = lane & 15;
  int half = lane >> 4;
  v8f z = {};
  ai = z; af = z; ag = z; ao = z;
#pragma unroll
  for (int r = 0; r < 8; ++r) {
    const float* gp = Grow + (size_t)(mt * 16 + half * 8 + r) * 1024 +
                      ht * 16 + nn;
    ai[r] = gp[0];
    af[r] = gp[256];
    ag[r] = gp[512];
    ao[r] = gp[768];
  }
#pragma unroll
  for (int k0 = 0; k0 < 256; k0 += 32) {
    v16h a = load_frag(hsh, mt * 16, 256, k0, lane);
    ai = wmma32(a, load_frag(Whh, 0 * 256 + ht * 16, 256, k0, lane), ai);
    af = wmma32(a, load_frag(Whh, 1 * 256 + ht * 16, 256, k0, lane), af);
    ag = wmma32(a, load_frag(Whh, 2 * 256 + ht * 16, 256, k0, lane), ag);
    ao = wmma32(a, load_frag(Whh, 3 * 256 + ht * 16, 256, k0, lane), ao);
  }
}

__device__ __forceinline__ void cell_update(v8f ai, v8f af, v8f ag, v8f ao,
                                            v8f& c, int mt, int ht, int lane,
                                            _Float16* __restrict__ hsh,
                                            float* __restrict__ hs_s) {
  int nn = lane & 15;
  int half = lane >> 4;
#pragma unroll
  for (int r = 0; r < 8; ++r) {
    float I = sigm(ai[r]);
    float F = sigm(af[r]);
    float Gv = tanhf(ag[r]);
    float O = sigm(ao[r]);
    float cc = F * c[r] + I * Gv;
    c[r] = cc;
    float hv = O * tanhf(cc);
    int m = mt * 16 + half * 8 + r;
    int col = ht * 16 + nn;
    hsh[m * 256 + col] = (_Float16)hv;
    hs_s[(size_t)m * 256 + col] = hv;
  }
}

__global__ __launch_bounds__(512) void k_lstm_rec(
    const float* __restrict__ G,    // [2][8192][1024]
    const _Float16* __restrict__ w16, // Whh_f @ 262144, Whh_b @ 3*262144
    float* __restrict__ hs)         // [2][S][B][H] f32 out
{
  __shared__ _Float16 hsh[32 * 256];  // h state, f16, 16 KB

  int dir = blockIdx.x;               // 0 = forward, 1 = backward
  const float* Gd = G + (size_t)dir * 8192 * 1024;
  const _Float16* Whh = w16 + (size_t)(dir ? 3 : 1) * 262144;
  float* hsd = hs + (size_t)dir * 8192 * 256;

  int tid = threadIdx.x;
  int lane = tid & 31;
  int w = tid >> 5;
  int mt = w >> 3;       // 0..1
  int ht0 = w & 7;       // owns ht0 and ht0+8

  for (int i = tid; i < 32 * 256; i += 512) hsh[i] = (_Float16)0.0f;
  __syncthreads();

  v8f c0 = {};
  v8f c1 = {};

  for (int step = 0; step < 256; ++step) {
    int s = dir ? (255 - step) : step;
    const float* Grow = Gd + (size_t)s * 32 * 1024;
    float* hs_s = hsd + (size_t)s * 32 * 256;

    v8f i0, f0, g0, o0, i1, f1, g1, o1;
    gate_mm(Grow, Whh, hsh, mt, ht0,     lane, i0, f0, g0, o0);
    gate_mm(Grow, Whh, hsh, mt, ht0 + 8, lane, i1, f1, g1, o1);
    __syncthreads();  // all waves done reading old h
    cell_update(i0, f0, g0, o0, c0, mt, ht0,     lane, hsh, hs_s);
    cell_update(i1, f1, g1, o1, c1, mt, ht0 + 8, lane, hsh, hs_s);
    __syncthreads();  // new h visible to all waves
  }
}

// ---- Kernel 5: output projection  out[b*S+s][t] = [hf|hb] . Wout[t] + bout -

__global__ void k_out_proj(const float* __restrict__ hf,   // [S][B][256]
                           const float* __restrict__ hb,   // [S][B][256]
                           const float* __restrict__ Wout, // [10][512]
                           const float* __restrict__ bout, // [10]
                           float* __restrict__ out)        // [8192][10]
{
  int idx = blockIdx.x * 256 + threadIdx.x;
  if (idx >= 8192 * 10) return;
  int t = idx % 10;
  int row = idx / 10;          // row = b*S + s (reference transpose order)
  int b = row >> 8;
  int s = row & 255;
  const float* hfp = hf + ((size_t)s * 32 + b) * 256;
  const float* hbp = hb + ((size_t)s * 32 + b) * 256;
  const float* wv = Wout + (size_t)t * 512;
  float acc = bout[t];
#pragma unroll 4
  for (int j = 0; j < 256; ++j)
    acc = fmaf(hfp[j], wv[j], fmaf(hbp[j], wv[256 + j], acc));
  out[idx] = acc;
}

// ---- Kernel 6: Viterbi over flattened sequence (L=8192, T=10) -------------

__global__ void k_viterbi(const float* __restrict__ feats, // [8192][10]
                          const float* __restrict__ trans, // [10][10] (to,from)
                          int* __restrict__ bptrs,         // [8192][10]
                          float* __restrict__ score_out,   // [1]
                          float* __restrict__ path_out)    // [8192] (as float)
{
  __shared__ float tr[128];
  __shared__ float fv[16];
  __shared__ float fvn[16];
  int lane = threadIdx.x;

  for (int i = lane; i < 100; i += 32) tr[i] = trans[i];
  if (lane < 16) fv[lane] = (lane == SOS_IDX) ? 0.0f : NEG_INF;
  __syncthreads();

  for (int l = 0; l < 8192; ++l) {
    if (lane < 10) {
      float best = NEG_INF * 4.0f;
      int bp = 0;
#pragma unroll
      for (int f = 0; f < 10; ++f) {
        float v = fv[f] + tr[lane * 10 + f];
        if (v > best) { best = v; bp = f; }
      }
      fvn[lane] = best + feats[(size_t)l * 10 + lane];
      bptrs[l * 10 + lane] = bp;
    }
    __syncthreads();
    if (lane < 10) fv[lane] = fvn[lane];
    __syncthreads();
  }

  if (lane == 0) {
    float best = NEG_INF * 4.0f;
    int best_last = 0;
#pragma unroll
    for (int t = 0; t < 10; ++t) {
      float v = fv[t] + tr[EOS_IDX * 10 + t];
      if (v > best) { best = v; best_last = t; }
    }
    score_out[0] = best;
    int tag = best_last;
    for (int l = 8191; l >= 0; --l) {
      path_out[l] = (float)tag;       // emit current tag, then follow bptr
      tag = bptrs[l * 10 + tag];
    }
  }
}

// ---------------------------------------------------------------------------

extern "C" void kernel_launch(void* const* d_in, const int* in_sizes, int n_in,
                              void* d_out, int out_size, void* d_ws,
                              size_t ws_size, hipStream_t stream) {
  (void)in_sizes; (void)n_in; (void)out_size; (void)ws_size;

  const int*   x      = (const int*)d_in[0];
  // d_in[1] = x_lengths (all S, unused), d_in[2] = y (unused by reference path)
  const float* emb    = (const float*)d_in[3];
  const float* Wih_f  = (const float*)d_in[4];
  const float* Whh_f  = (const float*)d_in[5];
  const float* bih_f  = (const float*)d_in[6];
  const float* bhh_f  = (const float*)d_in[7];
  const float* Wih_b  = (const float*)d_in[8];
  const float* Whh_b  = (const float*)d_in[9];
  const float* bih_b  = (const float*)d_in[10];
  const float* bhh_b  = (const float*)d_in[11];
  const float* Wout   = (const float*)d_in[12];
  const float* bout   = (const float*)d_in[13];
  const float* trans  = (const float*)d_in[14];

  // workspace carve-up (256B aligned)
  char* ws = (char*)d_ws;
  size_t off = 0;
  auto carve = [&](size_t bytes) {
    void* p = ws + off;
    off = (off + bytes + 255) & ~(size_t)255;
    return p;
  };
  _Float16* e16  = (_Float16*)carve((size_t)8192 * 256 * 2);       // 4 MB
  _Float16* w16  = (_Float16*)carve((size_t)4 * 262144 * 2);       // 2 MB
  float*    bias = (float*)carve((size_t)2 * 1024 * 4);            // 8 KB
  float*    G    = (float*)carve((size_t)2 * 8192 * 1024 * 4);     // 64 MB
  float*    hs   = (float*)carve((size_t)2 * 8192 * 256 * 4);      // 16 MB
  int*      bptr = (int*)carve((size_t)8192 * 10 * 4);             // 320 KB

  float* out_f   = (float*)d_out;            // [8192][10]
  float* score_f = out_f + 8192 * 10;        // [1]
  float* path_f  = score_f + 1;              // [8192]

  // 1) embedding gather + f16 cast
  k_embed_cast<<<8192, 256, 0, stream>>>(x, emb, e16);

  // 2) weight casts + fused biases
  k_cast_weights<<<4096, 256, 0, stream>>>(Wih_f, Whh_f, bih_f, bhh_f,
                                           Wih_b, Whh_b, bih_b, bhh_b,
                                           w16, bias);

  // 3) hoisted input GEMMs, both directions (WMMA)
  k_input_gemm<<<8192, 256, 0, stream>>>(e16, w16, bias, G);

  // 4) sequential recurrences: forward + backward CONCURRENTLY (2 blocks)
  k_lstm_rec<<<2, 512, 0, stream>>>(G, w16, hs);

  // 5) output projection straight into d_out
  float* hf = hs;
  float* hb = hs + (size_t)8192 * 256;
  k_out_proj<<<320, 256, 0, stream>>>(hf, hb, Wout, bout, out_f);

  // 6) Viterbi decode (score + path into d_out tail)
  k_viterbi<<<1, 32, 0, stream>>>(out_f, trans, bptr, score_f, path_f);
}
